// GLSLayer_23373212025557
// MI455X (gfx1250) — compile-verified, matrix operand order checked by reading, not run on hardware
//
#include <hip/hip_runtime.h>
#include <hip/hip_bf16.h>

typedef __attribute__((ext_vector_type(16))) _Float16 v16h;
typedef __attribute__((ext_vector_type(8)))  _Float16 v8h;
typedef __attribute__((ext_vector_type(8)))  float    v8f;

#define BDIM    4096
#define NFEAT   8192
#define NCLS    1000
#define NCLSPAD 1024
#define GLS_MAX_ITERS 10000
#define LDT 40                 // LDS row stride in halves (32 data + 8 pad = 80B)
#define TILEH (128 * LDT)      // halves per 128x32 tile in LDS

// ---------------------------------------------------------------------------
// Kernel 1: GLS firing counts. One thread per (row, feature) element.
// ---------------------------------------------------------------------------
__global__ __launch_bounds__(256) void gls_fire(const float* __restrict__ x,
                                                const float* __restrict__ b,
                                                const float* __restrict__ q,
                                                const float* __restrict__ e,
                                                _Float16* __restrict__ k16) {
  const int idx = blockIdx.x * 256 + threadIdx.x;
  const int n   = idx & (NFEAT - 1);
  const float xv = x[idx];
  const float bv = b[n];
  const float qv = q[n];
  const float ev = e[n];
  const float invb   = 1.0f / bv;
  const float inv1mb = 1.0f / (1.0f - bv);
  const float lo = xv - ev, hi = xv + ev;
  float m = qv;
  int k = 1;
  for (int it = 0; it < GLS_MAX_ITERS; ++it) {
    m = (m < bv) ? (m * invb) : ((1.0f - m) * inv1mb);
    if (m >= lo && m <= hi) break;
    ++k;
  }
  k16[idx] = (_Float16)(float)k;
}

// ---------------------------------------------------------------------------
// Kernel 2: normalize M rows -> f16; rows >= NCLS zero-padded.
// ---------------------------------------------------------------------------
__global__ __launch_bounds__(256) void gls_norm_m(const float* __restrict__ M,
                                                  _Float16* __restrict__ Mn) {
  const int c = blockIdx.x;  // 0..1023
  _Float16* out = Mn + (size_t)c * NFEAT;
  if (c >= NCLS) {
    for (int i = threadIdx.x; i < NFEAT; i += 256) out[i] = (_Float16)0.0f;
    return;
  }
  __shared__ float red[8];
  const float* row = M + (size_t)c * NFEAT;
  float ss = 0.0f;
  for (int i = threadIdx.x; i < NFEAT; i += 256) {
    float v = row[i];
    ss += v * v;
  }
  for (int off = 16; off > 0; off >>= 1) ss += __shfl_down(ss, off, 32);
  const int lane = threadIdx.x & 31, wv = threadIdx.x >> 5;
  if (lane == 0) red[wv] = ss;
  __syncthreads();
  if (wv == 0) {
    float s = (lane < 8) ? red[lane] : 0.0f;
    for (int off = 4; off > 0; off >>= 1) s += __shfl_down(s, off, 32);
    if (lane == 0) red[0] = s;
  }
  __syncthreads();
  const float scale = 1.0f / fmaxf(sqrtf(red[0]), 1e-8f);
  for (int i = threadIdx.x; i < NFEAT; i += 256) out[i] = (_Float16)(row[i] * scale);
}

// ---------------------------------------------------------------------------
// Async global -> LDS copy of one 16B segment per lane (ASYNCcnt tracked).
// ---------------------------------------------------------------------------
__device__ __forceinline__ void async_b128(unsigned lds_byte_addr,
                                           const _Float16* gaddr) {
  asm volatile("global_load_async_to_lds_b128 %0, %1, off"
               :: "v"(lds_byte_addr), "v"(gaddr)
               : "memory");
}
__device__ __forceinline__ void wait_async0() {
  asm volatile("s_wait_asynccnt 0x0" ::: "memory");
}

// ---------------------------------------------------------------------------
// Kernel 3: WMMA GEMM  D[4096][1024] = K16[4096][8192] * Mn[1024][8192]^T
// 128x128x32 block tile, 4 waves (2x2), each wave 64x64 = 4x4 WMMA accums.
// Double-buffered tiles filled with global_load_async_to_lds_b128.
// ---------------------------------------------------------------------------
__global__ __launch_bounds__(128) void gls_gemm(const _Float16* __restrict__ A,
                                                const _Float16* __restrict__ Bm,
                                                float* __restrict__ D) {
  __shared__ _Float16 lds[4 * TILEH];  // [buf0: A,B][buf1: A,B] = 40 KB

  const int tid   = threadIdx.x;
  const int wave  = tid >> 5;      // 0..3
  const int lane  = tid & 31;
  const int waveM = wave >> 1;     // 0..1
  const int waveN = wave & 1;      // 0..1

  const int blockM = blockIdx.x * 128;  // 32 tiles
  const int blockN = blockIdx.y * 128;  // 8 tiles (padded 1024 classes)

  // Thread t owns row t of both tiles: 4 async b128 segments each.
  const _Float16* gA = A  + (size_t)(blockM + tid) * NFEAT;
  const _Float16* gB = Bm + (size_t)(blockN + tid) * NFEAT;
  // LDS byte address = low 32 bits of generic pointer (aperture mapping).
  const unsigned ldsBase   = (unsigned)(size_t)(&lds[0]);
  const unsigned rowA      = ldsBase + (unsigned)tid * (LDT * 2);
  const unsigned bufStride = 2u * TILEH * sizeof(_Float16);
  const unsigned bOff      = TILEH * sizeof(_Float16);

  // WMMA 16-bit operand layout: lane<16 holds K 0..7 & 16..23 of row (lane&15),
  // lane>=16 holds K 8..15 & 24..31.
  const int laneRow  = lane & 15;
  const int laneHalf = (lane >> 4) * 8;

  v8f acc[4][4] = {};

  // Prime buffer 0.
#pragma unroll
  for (int s = 0; s < 4; ++s) {
    async_b128(rowA + s * 16, gA + s * 8);
    async_b128(rowA + bOff + s * 16, gB + s * 8);
  }
  wait_async0();
  __syncthreads();

  int buf = 0;
  for (int k0 = 0; k0 < NFEAT; k0 += 32) {
    // Issue async loads for the next K-slab into the other buffer.
    if (k0 + 32 < NFEAT) {
      const unsigned nb = (unsigned)(buf ^ 1) * bufStride;
#pragma unroll
      for (int s = 0; s < 4; ++s) {
        async_b128(rowA + nb + s * 16, gA + k0 + 32 + s * 8);
        async_b128(rowA + nb + bOff + s * 16, gB + k0 + 32 + s * 8);
      }
    }

    const _Float16* sA = lds + (size_t)buf * 2 * TILEH;
    const _Float16* sB = sA + TILEH;

    v16h afrag[4], bfrag[4];
#pragma unroll
    for (int f = 0; f < 4; ++f) {
      const int ra = (waveM * 64 + f * 16 + laneRow) * LDT + laneHalf;
      const v8h alo = *(const v8h*)&sA[ra];
      const v8h ahi = *(const v8h*)&sA[ra + 16];
      afrag[f] = __builtin_shufflevector(alo, ahi, 0, 1, 2, 3, 4, 5, 6, 7,
                                         8, 9, 10, 11, 12, 13, 14, 15);
      const int rb = (waveN * 64 + f * 16 + laneRow) * LDT + laneHalf;
      const v8h blo = *(const v8h*)&sB[rb];
      const v8h bhi = *(const v8h*)&sB[rb + 16];
      bfrag[f] = __builtin_shufflevector(blo, bhi, 0, 1, 2, 3, 4, 5, 6, 7,
                                         8, 9, 10, 11, 12, 13, 14, 15);
    }

#pragma unroll
    for (int i = 0; i < 4; ++i)
#pragma unroll
      for (int j = 0; j < 4; ++j)
        acc[i][j] = __builtin_amdgcn_wmma_f32_16x16x32_f16(
            /*neg_a=*/false, afrag[i], /*neg_b=*/false, bfrag[j],
            /*c_mod=*/(short)0, acc[i][j], /*reuse_a=*/false, /*reuse_b=*/false);

    wait_async0();       // next buffer's LDS writes complete (this wave)
    __syncthreads();     // visible to all waves; everyone done reading cur
    buf ^= 1;
  }

  // Epilogue. C/D layout: lane<16 -> N=lane, M=r; lane>=16 -> N=lane-16, M=r+8.
  const int nBase = blockN + waveN * 64 + (lane & 15);
  const int mBase = blockM + waveM * 64 + (lane >> 4) * 8;
#pragma unroll
  for (int i = 0; i < 4; ++i) {
#pragma unroll
    for (int j = 0; j < 4; ++j) {
      const int n = nBase + j * 16;
#pragma unroll
      for (int r = 0; r < 8; ++r) {
        const int m = mBase + i * 16 + r;
        D[(size_t)m * NCLSPAD + n] = acc[i][j][r];
      }
    }
  }
}

// ---------------------------------------------------------------------------
// Kernel 4: per-row argmax over NCLS with first-index tie-break (jnp.argmax).
// ---------------------------------------------------------------------------
__global__ __launch_bounds__(256) void gls_argmax(const float* __restrict__ sims,
                                                  int* __restrict__ out) {
  const int row = blockIdx.x;
  const float* s = sims + (size_t)row * NCLSPAD;
  float best = -3.402823466e38f;
  int bidx = 0x7fffffff;
  for (int i = threadIdx.x; i < NCLS; i += 256) {
    const float v = s[i];
    if (v > best || (v == best && i < bidx)) { best = v; bidx = i; }
  }
  __shared__ float bv[256];
  __shared__ int   bi[256];
  bv[threadIdx.x] = best;
  bi[threadIdx.x] = bidx;
  __syncthreads();
  for (int off = 128; off > 0; off >>= 1) {
    if (threadIdx.x < off) {
      const float v2 = bv[threadIdx.x + off];
      const int   i2 = bi[threadIdx.x + off];
      if (v2 > bv[threadIdx.x] ||
          (v2 == bv[threadIdx.x] && i2 < bi[threadIdx.x])) {
        bv[threadIdx.x] = v2;
        bi[threadIdx.x] = i2;
      }
    }
    __syncthreads();
  }
  if (threadIdx.x == 0) out[row] = bi[0];
}

// ---------------------------------------------------------------------------
extern "C" void kernel_launch(void* const* d_in, const int* in_sizes, int n_in,
                              void* d_out, int out_size, void* d_ws, size_t ws_size,
                              hipStream_t stream) {
  const float* x = (const float*)d_in[0];  // [4096, 8192]
  const float* b = (const float*)d_in[1];  // [8192]
  const float* q = (const float*)d_in[2];  // [8192]
  const float* e = (const float*)d_in[3];  // [8192]
  const float* M = (const float*)d_in[4];  // [1000, 8192]

  char* ws = (char*)d_ws;
  _Float16* k16 = (_Float16*)ws;                               // 67.1 MB
  size_t off = (size_t)BDIM * NFEAT * sizeof(_Float16);
  _Float16* Mn = (_Float16*)(ws + off);                        // 16.8 MB (padded 1024 rows)
  off += (size_t)NCLSPAD * NFEAT * sizeof(_Float16);
  float* sims = (float*)(ws + off);                            // 16.8 MB

  gls_fire<<<dim3((BDIM * (size_t)NFEAT) / 256), dim3(256), 0, stream>>>(x, b, q, e, k16);
  gls_norm_m<<<dim3(NCLSPAD), dim3(256), 0, stream>>>(M, Mn);
  gls_gemm<<<dim3(BDIM / 128, NCLSPAD / 128), dim3(256), 0, stream>>>(k16, Mn, sims);
  gls_argmax<<<dim3(BDIM), dim3(256), 0, stream>>>(sims, (int*)d_out);
}
